// AttentionBlock_43791486550430
// MI455X (gfx1250) — matrix operand order repaired and checked by reading, not compiled, not run on hardware
//
#include <hip/hip_runtime.h>

// ---------------------------------------------------------------------------
// Types for CDNA5 WMMA (gfx1250, wave32)
// ---------------------------------------------------------------------------
typedef __bf16 bf16;
typedef __attribute__((ext_vector_type(16))) __bf16 bf16x16;
typedef __attribute__((ext_vector_type(8)))  float  f32x8;

#define NHEADS   12
#define NGROUPS  32
#define C_DIM    768
#define T_DIM    1024
#define S_ENC    256
#define S_TOT    1280      // 256 encoder keys + 1024 self keys
#define CH       64        // head dim
#define BATCH    8
#define K_DIM    768       // contraction dim of all projection GEMMs

// D = A(16x32 bf16) * B(32x16 bf16) + C(16x16 f32)
static __device__ __forceinline__ f32x8 wmma_bf16(bf16x16 a, bf16x16 b, f32x8 c) {
  return __builtin_amdgcn_wmma_f32_16x16x32_bf16(false, a, false, b, (short)0, c,
                                                 false, false);
}

// ---------------------------------------------------------------------------
// Fragment loaders, lane mappings per CDNA5 ISA 7.12.2 (wave32).
// A 16x32 (MxK): lane l -> m = l%16, h = l/16; element e -> k = 8h+e (e<8),
//                k = 16+8h+(e-8) (e>=8). Runs are contiguous in K, so a
//                row-major [M][K] source gives two 16B vector loads.
// B 32x16 (KxN): lane l -> n = l%16, h = l/16; element e -> same k as A.
//                The e-run walks K, so we keep all B operands TRANSPOSED in
//                memory ([N][K]) and load two contiguous 16B vectors.
// C/D 16x16:     VGPR r -> m = r + 8*(l/16), n = l%16.
// All leading dims (64/768/1280 elems) are multiples of 16B; K offsets are
// multiples of 16B -> uint4 loads are aligned.
// ---------------------------------------------------------------------------
static __device__ __forceinline__
bf16x16 load_a_frag(const bf16* __restrict__ src, int lda, int m0, int k0, int lane) {
  const int m = m0 + (lane & 15);
  const int h = lane >> 4;
  const bf16* p = src + (size_t)m * lda + k0 + 8 * h;
  union { uint4 u[2]; bf16x16 v; } f;
  f.u[0] = *(const uint4*)(p);        // k = k0 + 8h + (0..7)
  f.u[1] = *(const uint4*)(p + 16);   // k = k0 + 16 + 8h + (0..7)
  return f.v;
}

// src is the TRANSPOSE of the logical B: src[n][k], leading dim ldk (K extent).
static __device__ __forceinline__
bf16x16 load_bt_frag(const bf16* __restrict__ src, int ldk, int k0, int n0, int lane) {
  const int n = n0 + (lane & 15);
  const int h = lane >> 4;
  const bf16* p = src + (size_t)n * ldk + k0 + 8 * h;
  union { uint4 u[2]; bf16x16 v; } f;
  f.u[0] = *(const uint4*)(p);        // k = k0 + 8h + (0..7)
  f.u[1] = *(const uint4*)(p + 16);   // k = k0 + 16 + 8h + (0..7)
  return f.v;
}

// ---------------------------------------------------------------------------
// fp32 -> bf16 conversion (weights stay row-major [M][K])
// ---------------------------------------------------------------------------
__global__ void f32_to_bf16_kernel(const float* __restrict__ src,
                                   bf16* __restrict__ dst, int n) {
  int i = blockIdx.x * blockDim.x + threadIdx.x;
  if (i < n) dst[i] = (bf16)src[i];
}

// src [B][rows][cols] fp32 -> dst [B][cols][rows] bf16 (per-batch transpose)
__global__ void transpose_to_bf16_kernel(const float* __restrict__ src,
                                         bf16* __restrict__ dst,
                                         int rows, int cols, int total) {
  int i = blockIdx.x * blockDim.x + threadIdx.x;
  if (i >= total) return;
  int per = rows * cols;
  int b = i / per;
  int r = (i % per) / cols;
  int c = i % cols;
  dst[((size_t)b * cols + c) * rows + r] = (bf16)src[i];
}

// ---------------------------------------------------------------------------
// GroupNorm: one block per (batch, group); 24 channels * 1024 tokens.
// Emits normalized activations as bf16 in TRANSPOSED layout xn_t[B][T][C]
// (the GEMM B-operand layout).
// ---------------------------------------------------------------------------
__global__ __launch_bounds__(256)
void groupnorm_kernel(const float* __restrict__ x, const float* __restrict__ gw,
                      const float* __restrict__ gb, bf16* __restrict__ xn_t) {
  const int tid = threadIdx.x;
  const int g = blockIdx.x & (NGROUPS - 1);
  const int b = blockIdx.x >> 5;
  const int CPG = C_DIM / NGROUPS;   // 24
  const int NEL = CPG * T_DIM;       // 24576
  const float* xp = x + ((size_t)b * C_DIM + g * CPG) * T_DIM;

  float s = 0.f, ss = 0.f;
  for (int i = tid; i < NEL; i += 256) { float v = xp[i]; s += v; ss += v * v; }

  __shared__ float red[256];
  __shared__ float stats[2];
  red[tid] = s; __syncthreads();
  for (int off = 128; off > 0; off >>= 1) {
    if (tid < off) red[tid] += red[tid + off];
    __syncthreads();
  }
  if (tid == 0) stats[0] = red[0];
  __syncthreads();
  red[tid] = ss; __syncthreads();
  for (int off = 128; off > 0; off >>= 1) {
    if (tid < off) red[tid] += red[tid + off];
    __syncthreads();
  }
  if (tid == 0) stats[1] = red[0];
  __syncthreads();

  const float mean = stats[0] / (float)NEL;
  const float var  = stats[1] / (float)NEL - mean * mean;
  const float inv  = rsqrtf(var + 1e-5f);
  // Write pass iterates c-fastest so the transposed stores are coalesced.
  for (int j = tid; j < NEL; j += 256) {
    int cl = j % CPG;
    int t  = j / CPG;
    int c  = g * CPG + cl;
    float v = (xp[(size_t)cl * T_DIM + t] - mean) * inv * gw[c] + gb[c];
    xn_t[((size_t)b * T_DIM + t) * C_DIM + c] = (bf16)v;
  }
}

// ---------------------------------------------------------------------------
// Generic WMMA GEMM:  out[M][N] = A[M][K_DIM] * Bt[b][N][K_DIM]^T
// Block: 256 threads = 8 waves in a 4(M) x 2(N) arrangement -> 128x128 tile.
// Wave: 32(M) x 64(N) = 2x4 accumulator tiles, K stepped by 32.
// Epilogue functor handles bias/scale/scatter per use-site.
// ---------------------------------------------------------------------------
template <typename Epi>
__global__ __launch_bounds__(256)
void gemm_kernel(const bf16* __restrict__ A, const bf16* __restrict__ Bt,
                 int N, Epi epi) {
  const int lane = threadIdx.x & 31;
  const int w    = threadIdx.x >> 5;
  const int bm0  = blockIdx.y * 128 + (w >> 1) * 32;
  const int bn0  = blockIdx.x * 128 + (w & 1) * 64;
  const int b    = blockIdx.z;
  const bf16* Bb = Bt + (size_t)b * N * K_DIM;

  f32x8 acc[2][4] = {};
  for (int k0 = 0; k0 < K_DIM; k0 += 32) {
    if (k0 + 32 < K_DIM) {  // global_prefetch_b8 for the next A K-slab
      __builtin_prefetch(A + (size_t)(bm0 + (lane & 15)) * K_DIM + k0 + 32, 0, 1);
    }
    bf16x16 a0 = load_a_frag(A, K_DIM, bm0,      k0, lane);
    bf16x16 a1 = load_a_frag(A, K_DIM, bm0 + 16, k0, lane);
#pragma unroll
    for (int j = 0; j < 4; ++j) {
      bf16x16 bfr = load_bt_frag(Bb, K_DIM, k0, bn0 + 16 * j, lane);
      acc[0][j] = wmma_bf16(a0, bfr, acc[0][j]);
      acc[1][j] = wmma_bf16(a1, bfr, acc[1][j]);
    }
  }
#pragma unroll
  for (int mt = 0; mt < 2; ++mt)
#pragma unroll
    for (int j = 0; j < 4; ++j)
#pragma unroll
      for (int r = 0; r < 8; ++r) {
        int o = bm0 + mt * 16 + r + 8 * (lane >> 4);
        int n = bn0 + j * 16 + (lane & 15);
        epi.store(acc[mt][j][r], o, n, b);
      }
}

// Epilogues -----------------------------------------------------------------
// Attention operand layouts (chosen so every WMMA B-fragment is contiguous):
//   Q  t-major [hb][T][CH]   (A-matrix of S = q^T k, K-run walks c)
//   K  s-major [hb][S][CH]   (S-GEMM B walks c -> c contiguous)
//   V  c-major [hb][CH][S]   (O-GEMM B walks s -> s contiguous)
// q,k pre-scaled by ch^-0.25.
struct QkvEpi {
  const float* bias; bf16* Q; bf16* K; bf16* V; float scale;
  __device__ void store(float acc, int o, int t, int b) const {
    acc += bias[o];
    int head = o / 192, r = o % 192;
    size_t hb = (size_t)b * NHEADS + head;
    if (r < 64)
      Q[(hb * T_DIM + t) * CH + r] = (bf16)(acc * scale);
    else if (r < 128)
      K[(hb * S_TOT + S_ENC + t) * CH + (r - 64)] = (bf16)(acc * scale);
    else
      V[(hb * CH + (r - 128)) * S_TOT + S_ENC + t] = (bf16)acc;
  }
};

struct EkvEpi {
  const float* bias; bf16* K; bf16* V; float scale;
  __device__ void store(float acc, int o, int s, int b) const {
    acc += bias[o];
    int head = o / 128, r = o % 128;
    size_t hb = (size_t)b * NHEADS + head;
    if (r < 64)
      K[(hb * S_TOT + s) * CH + r] = (bf16)(acc * scale);
    else
      V[(hb * CH + (r - 64)) * S_TOT + s] = (bf16)acc;
  }
};

struct ProjEpi {
  const float* bias; const float* xres; float* out;
  __device__ void store(float acc, int o, int t, int b) const {
    size_t idx = ((size_t)b * C_DIM + o) * T_DIM + t;
    out[idx] = acc + bias[o] + xres[idx];
  }
};

// ---------------------------------------------------------------------------
// Flash attention: one wave per 16-query tile; stream S_TOT keys in 32-chunks.
// S = q^T k (K=64 via 2 chained WMMAs), online softmax with shfl_xor row
// reductions (row m = r + 8*(lane/16) lives in one 16-lane half), P goes
// through LDS to be re-read in A-fragment layout, then O += P * V.
// Output written bf16 to a_t[B][T][C] (B-operand layout for the projection).
// ---------------------------------------------------------------------------
__global__ __launch_bounds__(256)
void attention_kernel(const bf16* __restrict__ Q, const bf16* __restrict__ K,
                      const bf16* __restrict__ V, bf16* __restrict__ Aout) {
  const int lane = threadIdx.x & 31;
  const int w    = threadIdx.x >> 5;
  const int hb   = blockIdx.y;
  const int b    = hb / NHEADS;
  const int head = hb % NHEADS;
  const int t0   = (blockIdx.x * 8 + w) * 16;
  const int h    = lane >> 4;
  const int nl   = lane & 15;

  __shared__ bf16 Plds[8 * 16 * 32];     // 1KB per wave, private per wave
  bf16* pl = &Plds[w * 512];

  const bf16* Qp = Q + (size_t)hb * T_DIM * CH;
  const bf16* Kp = K + (size_t)hb * S_TOT * CH;   // s-major [S][CH]
  const bf16* Vp = V + (size_t)hb * CH * S_TOT;   // c-major [CH][S]

  const bf16x16 qa0 = load_a_frag(Qp, CH, t0, 0,  lane);  // c = 0..31
  const bf16x16 qa1 = load_a_frag(Qp, CH, t0, 32, lane);  // c = 32..63

  f32x8 O[4] = {};
  float rmax[8], rsum[8];
#pragma unroll
  for (int r = 0; r < 8; ++r) { rmax[r] = -3.0e38f; rsum[r] = 0.f; }

  for (int sA = 0; sA < S_TOT; sA += 32) {
    // ---- S tiles: 16 queries x 32 keys (B walks c; K is [s][c]) ------------
    f32x8 S0 = {}, S1 = {};
    {
      bf16x16 bk;
      bk = load_bt_frag(Kp, CH, 0,  sA,      lane); S0 = wmma_bf16(qa0, bk, S0);
      bk = load_bt_frag(Kp, CH, 32, sA,      lane); S0 = wmma_bf16(qa1, bk, S0);
      bk = load_bt_frag(Kp, CH, 0,  sA + 16, lane); S1 = wmma_bf16(qa0, bk, S1);
      bk = load_bt_frag(Kp, CH, 32, sA + 16, lane); S1 = wmma_bf16(qa1, bk, S1);
    }
    // ---- online softmax: row max / correction ------------------------------
    float corr[8];
#pragma unroll
    for (int r = 0; r < 8; ++r) {
      float m = fmaxf(S0[r], S1[r]);
      m = fmaxf(m, __shfl_xor(m, 1, 32));
      m = fmaxf(m, __shfl_xor(m, 2, 32));
      m = fmaxf(m, __shfl_xor(m, 4, 32));
      m = fmaxf(m, __shfl_xor(m, 8, 32));
      float nm = fmaxf(rmax[r], m);
      corr[r] = __expf(rmax[r] - nm);
      rmax[r] = nm;
    }
    // ---- P = exp(S - max), stash to LDS (row-major 16x32), row sums --------
#pragma unroll
    for (int r = 0; r < 8; ++r) {
      float p0 = __expf(S0[r] - rmax[r]);
      float p1 = __expf(S1[r] - rmax[r]);
      int row = r + 8 * h;
      pl[row * 32 + nl]      = (bf16)p0;
      pl[row * 32 + 16 + nl] = (bf16)p1;
      float rs = p0 + p1;
      rs += __shfl_xor(rs, 1, 32);
      rs += __shfl_xor(rs, 2, 32);
      rs += __shfl_xor(rs, 4, 32);
      rs += __shfl_xor(rs, 8, 32);
      rsum[r] = rsum[r] * corr[r] + rs;
    }
#pragma unroll
    for (int ct = 0; ct < 4; ++ct)
#pragma unroll
      for (int r = 0; r < 8; ++r) O[ct][r] *= corr[r];

    __builtin_amdgcn_wave_barrier();   // order intra-wave LDS write -> read
    // ---- O += P(16x32) * V(32x64)  (B walks s; V is [c][s]) ----------------
    bf16x16 pa = load_a_frag(pl, 32, 0, 0, lane);
#pragma unroll
    for (int ct = 0; ct < 4; ++ct) {
      bf16x16 bv = load_bt_frag(Vp, S_TOT, sA, ct * 16, lane);
      O[ct] = wmma_bf16(pa, bv, O[ct]);
    }
    __builtin_amdgcn_wave_barrier();   // keep next iter's stores after reads
  }

  // ---- normalize and scatter to a_t[B][T][C] -------------------------------
#pragma unroll
  for (int ct = 0; ct < 4; ++ct)
#pragma unroll
    for (int r = 0; r < 8; ++r) {
      float v = O[ct][r] / rsum[r];
      int t = t0 + r + 8 * h;
      int c = head * CH + ct * 16 + nl;
      Aout[((size_t)b * T_DIM + t) * C_DIM + c] = (bf16)v;
    }
}

// ---------------------------------------------------------------------------
// Host launcher
// ---------------------------------------------------------------------------
extern "C" void kernel_launch(void* const* d_in, const int* in_sizes, int n_in,
                              void* d_out, int out_size, void* d_ws, size_t ws_size,
                              hipStream_t stream) {
  (void)in_sizes; (void)n_in; (void)out_size; (void)ws_size;
  const float* x      = (const float*)d_in[0];
  const float* enc    = (const float*)d_in[1];
  const float* gn_w   = (const float*)d_in[2];
  const float* gn_b   = (const float*)d_in[3];
  const float* qkv_w  = (const float*)d_in[4];
  const float* qkv_b  = (const float*)d_in[5];
  const float* enc_w  = (const float*)d_in[6];
  const float* enc_b  = (const float*)d_in[7];
  const float* proj_w = (const float*)d_in[8];
  const float* proj_b = (const float*)d_in[9];
  float* out = (float*)d_out;

  // Workspace carve-out (bf16 buffers, 256B aligned), ~80 MB total.
  char* ws = (char*)d_ws;
  size_t off = 0;
  auto carve = [&](size_t elems) -> bf16* {
    bf16* p = (bf16*)(ws + off);
    off = (off + elems * sizeof(bf16) + 255) & ~(size_t)255;
    return p;
  };
  const size_t n_qkvw  = (size_t)3 * C_DIM * C_DIM;   // 2304x768
  const size_t n_encw  = (size_t)2 * C_DIM * C_DIM;   // 1536x768
  const size_t n_projw = (size_t)C_DIM * C_DIM;       // 768x768
  const size_t n_enc   = (size_t)BATCH * C_DIM * S_ENC;
  const size_t n_xn    = (size_t)BATCH * C_DIM * T_DIM;
  const size_t n_q     = (size_t)BATCH * NHEADS * T_DIM * CH;
  const size_t n_kv    = (size_t)BATCH * NHEADS * CH * S_TOT;

  bf16* qkv_w_bf  = carve(n_qkvw);
  bf16* enc_w_bf  = carve(n_encw);
  bf16* proj_w_bf = carve(n_projw);
  bf16* enc_t_bf  = carve(n_enc);   // [B][S][C]
  bf16* xn_t_bf   = carve(n_xn);    // [B][T][C]
  bf16* Qb        = carve(n_q);     // [hb][T][CH]
  bf16* Kb        = carve(n_kv);    // [hb][S][CH]
  bf16* Vb        = carve(n_kv);    // [hb][CH][S]
  bf16* At        = carve(n_xn);    // [B][T][C]

  const float scale = 0.35355339059327373f;  // 64^-0.25

  // 1) fp32 -> bf16 conversions (weights row-major; encoder_out transposed)
  f32_to_bf16_kernel<<<(int)((n_qkvw + 255) / 256), 256, 0, stream>>>(qkv_w, qkv_w_bf, (int)n_qkvw);
  f32_to_bf16_kernel<<<(int)((n_encw + 255) / 256), 256, 0, stream>>>(enc_w, enc_w_bf, (int)n_encw);
  f32_to_bf16_kernel<<<(int)((n_projw + 255) / 256), 256, 0, stream>>>(proj_w, proj_w_bf, (int)n_projw);
  transpose_to_bf16_kernel<<<(int)((n_enc + 255) / 256), 256, 0, stream>>>(
      enc, enc_t_bf, C_DIM, S_ENC, (int)n_enc);

  // 2) GroupNorm -> xn_t[B][T][C]
  groupnorm_kernel<<<BATCH * NGROUPS, 256, 0, stream>>>(x, gn_w, gn_b, xn_t_bf);

  // 3) QKV projection: M=2304, N=1024 -> grid (8, 18, 8)
  gemm_kernel<QkvEpi><<<dim3(T_DIM / 128, (3 * C_DIM) / 128, BATCH), 256, 0, stream>>>(
      qkv_w_bf, xn_t_bf, T_DIM, QkvEpi{qkv_b, Qb, Kb, Vb, scale});

  // 4) Encoder KV projection: M=1536, N=256 -> grid (2, 12, 8)
  gemm_kernel<EkvEpi><<<dim3(S_ENC / 128, (2 * C_DIM) / 128, BATCH), 256, 0, stream>>>(
      enc_w_bf, enc_t_bf, S_ENC, EkvEpi{enc_b, Kb, Vb, scale});

  // 5) Attention: 64 query tiles / 8 waves per block -> grid (8, 96)
  attention_kernel<<<dim3(T_DIM / (16 * 8), BATCH * NHEADS), 256, 0, stream>>>(
      Qb, Kb, Vb, At);

  // 6) Output projection + residual: M=768, N=1024 -> grid (8, 6, 8)
  gemm_kernel<ProjEpi><<<dim3(T_DIM / 128, C_DIM / 128, BATCH), 256, 0, stream>>>(
      proj_w_bf, At, T_DIM, ProjEpi{proj_b, x, out});
}
